// PHFormerTiny_51316269252745
// MI455X (gfx1250) — matrix operand order repaired and checked by reading, not compiled, not run on hardware
//
#include <hip/hip_runtime.h>
#include <hip/hip_bf16.h>
#include <math.h>

typedef __attribute__((ext_vector_type(16))) _Float16 v16h;
typedef __attribute__((ext_vector_type(8)))  _Float16 v8h;
typedef __attribute__((ext_vector_type(8)))  float    v8f;

__device__ __forceinline__ float gelu_f(float x) {
    return 0.5f * x * (1.0f + erff(x * 0.7071067811865475f));
}

enum { EPI_F16 = 0, EPI_F16_GELU = 1, EPI_RES_F32 = 2, EPI_EMBED = 3 };

union V16 { v16h v; v8h h[2]; };

// ---------------------------------------------------------------------------
// WMMA GEMM: C[M,N] = A[M,K](f16) * B[K,N](f32 weights, converted to f16)
// Block tile 128x128, 8 wave32 waves in 4(M)x2(N), each wave 32x64 = 2x4 WMMA.
// ---------------------------------------------------------------------------
template<int EPI>
__global__ __launch_bounds__(256)
void gemm_wmma_kernel(const _Float16* __restrict__ A,
                      const float* __restrict__ Bw,
                      const float* __restrict__ bias,
                      void* __restrict__ OutP,
                      float* __restrict__ Res,
                      const float* __restrict__ pos,
                      int M, int N, int K)
{
    __shared__ _Float16 As[128 * 40];   // [row][k], stride 40 halves (16B aligned)
    __shared__ _Float16 Bs[128 * 40];   // [col][k]

    const int tid  = threadIdx.x;
    const int lane = tid & 31;
    const int w    = tid >> 5;
    const int wm   = w & 3;          // wave M index 0..3
    const int wn   = w >> 2;         // wave N index 0..1
    const int m0   = blockIdx.y * 128;
    const int n0   = blockIdx.x * 128;

    const v8f vz = {0.f,0.f,0.f,0.f,0.f,0.f,0.f,0.f};
    v8f acc[2][4];
    #pragma unroll
    for (int i = 0; i < 2; ++i)
        #pragma unroll
        for (int j = 0; j < 4; ++j) acc[i][j] = vz;

    const int ar  = tid >> 1;            // A staging: row 0..127
    const int ac0 = (tid & 1) * 16;      // A staging: k offset 0 or 16
    const int bn  = tid >> 1;            // B staging: col 0..127
    const int bk0 = (tid & 1) * 16;

    for (int k0 = 0; k0 < K; k0 += 32) {
        // ---- stage A tile (f16 global, row-major lda=K) ----
        {
            const int gm = m0 + ar;
            _Float16* dst = &As[ar * 40 + ac0];
            if (gm < M) {
                const _Float16* src = A + (size_t)gm * K + k0 + ac0;
                *(v8h*)dst       = *(const v8h*)src;
                *(v8h*)(dst + 8) = *(const v8h*)(src + 8);
                if (k0 + 32 < K) __builtin_prefetch(src + 32, 0, 1);
            } else {
                const v8h hz = { (_Float16)0, (_Float16)0, (_Float16)0, (_Float16)0,
                                 (_Float16)0, (_Float16)0, (_Float16)0, (_Float16)0 };
                *(v8h*)dst = hz; *(v8h*)(dst + 8) = hz;
            }
        }
        // ---- stage B tile: weights f32 [K][N] -> LDS [n][k] f16 ----
        {
            const float* src = Bw + (size_t)(k0 + bk0) * N + (n0 + bn);
            _Float16* dst = &Bs[bn * 40 + bk0];
            #pragma unroll
            for (int kk = 0; kk < 16; ++kk)
                dst[kk] = (_Float16)src[(size_t)kk * N];
            if (k0 + 32 < K) __builtin_prefetch(src + (size_t)32 * N, 0, 1);
        }
        __syncthreads();

        // ---- build fragments (A 16x32 layout, B 32x16 layout per ISA) ----
        V16 af[2], bf[4];
        const int arow0 = wm * 32 + (lane & 15);
        const int klo   = (lane & 16) ? 8 : 0;      // lanes 16-31 hold K+8 / K+24
        #pragma unroll
        for (int i = 0; i < 2; ++i) {
            const _Float16* p = &As[(arow0 + i * 16) * 40 + klo];
            af[i].h[0] = *(const v8h*)p;            // K = klo .. klo+7
            af[i].h[1] = *(const v8h*)(p + 16);     // K = klo+16 .. klo+23
        }
        const int bcol0 = wn * 64 + (lane & 15);
        const int kb    = (lane & 16) ? 8 : 0;      // lanes 16-31 hold K+8 / K+24
        #pragma unroll
        for (int j = 0; j < 4; ++j) {
            const _Float16* p = &Bs[(bcol0 + j * 16) * 40 + kb];
            bf[j].h[0] = *(const v8h*)p;
            bf[j].h[1] = *(const v8h*)(p + 16);
        }
        #pragma unroll
        for (int i = 0; i < 2; ++i)
            #pragma unroll
            for (int j = 0; j < 4; ++j)
                acc[i][j] = __builtin_amdgcn_wmma_f32_16x16x32_f16(
                    false, af[i].v, false, bf[j].v, (short)0, acc[i][j], false, false);

        __syncthreads();
    }

    // ---- epilogue: C layout lanes 0-15 -> M=r, lanes 16-31 -> M=8+r ----
    const int lcol  = lane & 15;
    const int lhalf = lane >> 4;
    #pragma unroll
    for (int i = 0; i < 2; ++i) {
        #pragma unroll
        for (int j = 0; j < 4; ++j) {
            const int col = n0 + wn * 64 + j * 16 + lcol;
            const float bv = bias ? bias[col] : 0.f;
            #pragma unroll
            for (int r = 0; r < 8; ++r) {
                const int row = m0 + wm * 32 + i * 16 + r + 8 * lhalf;
                if (row >= M) continue;
                const float v = acc[i][j][r] + bv;
                if (EPI == EPI_F16) {
                    ((_Float16*)OutP)[(size_t)row * N + col] = (_Float16)v;
                } else if (EPI == EPI_F16_GELU) {
                    ((_Float16*)OutP)[(size_t)row * N + col] = (_Float16)gelu_f(v);
                } else if (EPI == EPI_RES_F32) {
                    Res[(size_t)row * N + col] += v;
                } else { // EPI_EMBED: input row (b,t of 511) -> x row b*512 + t + 1
                    const int b = row / 511;
                    const int t = row % 511;
                    const size_t orow = (size_t)b * 512 + t + 1;
                    ((float*)OutP)[orow * N + col] = v + pos[(size_t)(t + 1) * N + col];
                }
            }
        }
    }
}

// ---------------------------------------------------------------------------
// Row-wise LayerNorm (fp32 in) -> f16 out
// ---------------------------------------------------------------------------
__global__ __launch_bounds__(128)
void ln_rows_kernel(const float* __restrict__ X, const float* __restrict__ w,
                    const float* __restrict__ b, _Float16* __restrict__ O)
{
    const int row = blockIdx.x;
    const int tid = threadIdx.x;
    const float* xr = X + (size_t)row * 1024;
    __shared__ float red[128];
    float s = 0.f;
    for (int c = tid; c < 1024; c += 128) s += xr[c];
    red[tid] = s; __syncthreads();
    for (int st = 64; st > 0; st >>= 1) { if (tid < st) red[tid] += red[tid + st]; __syncthreads(); }
    const float mean = red[0] * (1.f / 1024.f);
    __syncthreads();
    float v = 0.f;
    for (int c = tid; c < 1024; c += 128) { const float d = xr[c] - mean; v += d * d; }
    red[tid] = v; __syncthreads();
    for (int st = 64; st > 0; st >>= 1) { if (tid < st) red[tid] += red[tid + st]; __syncthreads(); }
    const float inv = rsqrtf(red[0] * (1.f / 1024.f) + 1e-5f);
    _Float16* orow = O + (size_t)row * 1024;
    for (int c = tid; c < 1024; c += 128)
        orow[c] = (_Float16)((xr[c] - mean) * inv * w[c] + b[c]);
}

// ---------------------------------------------------------------------------
// qp = elu(q @ R) + 1, kp = elu(k @ R) + 1    (elu+1 == x>0 ? x+1 : exp(x))
// ---------------------------------------------------------------------------
__global__ __launch_bounds__(256)
void qpkp_kernel(const _Float16* __restrict__ qkv, const float* __restrict__ R,
                 _Float16* __restrict__ qp, _Float16* __restrict__ kp)
{
    const int row = blockIdx.x;   // b*512 + t
    const int tid = threadIdx.x;
    __shared__ _Float16 qs[1024];
    __shared__ _Float16 ks[1024];
    __shared__ float Rs[64 * 32];
    const _Float16* src = qkv + (size_t)row * 3072;
    for (int c = tid; c < 1024; c += 256) { qs[c] = src[c]; ks[c] = src[1024 + c]; }
    for (int c = tid; c < 2048; c += 256) Rs[c] = R[c];
    __syncthreads();
    #pragma unroll
    for (int e = 0; e < 2; ++e) {
        const int o = tid * 2 + e;          // o = h*32 + f
        const int h = o >> 5, f = o & 31;
        float aq = 0.f, ak = 0.f;
        for (int d = 0; d < 64; ++d) {
            const float r = Rs[d * 32 + f];
            aq += (float)qs[h * 64 + d] * r;
            ak += (float)ks[h * 64 + d] * r;
        }
        qp[(size_t)row * 512 + o] = (_Float16)(aq > 0.f ? aq + 1.f : __expf(aq));
        kp[(size_t)row * 512 + o] = (_Float16)(ak > 0.f ? ak + 1.f : __expf(ak));
    }
}

// ---------------------------------------------------------------------------
// kv[b,h,f,d] = sum_t kp[b,t,h,f]*v[b,t,h,d];  ksum[b,h,f] = sum_t kp
// ---------------------------------------------------------------------------
__global__ __launch_bounds__(256)
void kvsum_kernel(const _Float16* __restrict__ kp, const _Float16* __restrict__ qkv,
                  float* __restrict__ kv, float* __restrict__ ksum)
{
    const int bh = blockIdx.x;
    const int b = bh >> 4, h = bh & 15;
    const int tid = threadIdx.x;
    __shared__ _Float16 kps[64 * 32];
    __shared__ _Float16 vs[64 * 64];
    float acc[8];
    #pragma unroll
    for (int i = 0; i < 8; ++i) acc[i] = 0.f;
    float ks = 0.f;
    for (int tc = 0; tc < 512; tc += 64) {
        #pragma unroll
        for (int i = 0; i < 8; ++i) {
            const int idx = i * 256 + tid;
            const int tt = idx >> 5, f = idx & 31;
            kps[idx] = kp[((size_t)(b * 512 + tc + tt)) * 512 + h * 32 + f];
        }
        #pragma unroll
        for (int i = 0; i < 16; ++i) {
            const int idx = i * 256 + tid;
            const int tt = idx >> 6, d = idx & 63;
            vs[idx] = qkv[((size_t)(b * 512 + tc + tt)) * 3072 + 2048 + h * 64 + d];
        }
        __syncthreads();
        #pragma unroll
        for (int i = 0; i < 8; ++i) {
            const int o = i * 256 + tid;
            const int f = o >> 6, d = o & 63;
            float a = acc[i];
            for (int tt = 0; tt < 64; ++tt)
                a += (float)kps[tt * 32 + f] * (float)vs[tt * 64 + d];
            acc[i] = a;
        }
        if (tid < 32)
            for (int tt = 0; tt < 64; ++tt) ks += (float)kps[tt * 32 + tid];
        __syncthreads();
    }
    #pragma unroll
    for (int i = 0; i < 8; ++i) kv[(size_t)bh * 2048 + i * 256 + tid] = acc[i];
    if (tid < 32) ksum[bh * 32 + tid] = ks;
}

// ---------------------------------------------------------------------------
// z[b,t,h,d] = (sum_f qp*kv) / (sum_f qp*ksum + 1e-6) -> f16
// ---------------------------------------------------------------------------
__global__ __launch_bounds__(256)
void z_kernel(const _Float16* __restrict__ qp, const float* __restrict__ kv,
              const float* __restrict__ ksum, _Float16* __restrict__ zout)
{
    const int blk = blockIdx.x;               // b*64 + h*4 + chunk
    const int chunk = blk & 3;
    const int h = (blk >> 2) & 15;
    const int b = blk >> 6;
    const int tid = threadIdx.x;
    __shared__ float kvs[2048];
    __shared__ float kss[32];
    __shared__ _Float16 qps[128 * 32];
    for (int i = tid; i < 2048; i += 256) kvs[i] = kv[(size_t)(b * 16 + h) * 2048 + i];
    if (tid < 32) kss[tid] = ksum[(b * 16 + h) * 32 + tid];
    const int rbase = b * 512 + chunk * 128;
    for (int i = tid; i < 128 * 32; i += 256) {
        const int tl = i >> 5, f = i & 31;
        qps[i] = qp[(size_t)(rbase + tl) * 512 + h * 32 + f];
    }
    __syncthreads();
    const int tty = tid >> 6, d = tid & 63;
    for (int ti = 0; ti < 32; ++ti) {
        const int tl = ti * 4 + tty;
        float z = 0.f, den = 0.f;
        #pragma unroll
        for (int f = 0; f < 32; ++f) {
            const float q = (float)qps[tl * 32 + f];
            z += q * kvs[f * 64 + d];
            den += q * kss[f];
        }
        zout[(size_t)(rbase + tl) * 1024 + h * 64 + d] = (_Float16)(z / (den + 1e-6f));
    }
}

// ---------------------------------------------------------------------------
// Topo features + 1024x1024 matvec -> x row 0 (+pos_embed row 0)
// ---------------------------------------------------------------------------
__global__ __launch_bounds__(256)
void topo_kernel(const float* __restrict__ pd, const int* __restrict__ betti,
                 const float* __restrict__ pim,
                 const float* __restrict__ Wpd1, const float* __restrict__ bpd1,
                 const float* __restrict__ lnpdw, const float* __restrict__ lnpdb,
                 const float* __restrict__ Wpd2, const float* __restrict__ bpd2,
                 const float* __restrict__ btab,
                 const float* __restrict__ convw, const float* __restrict__ convb,
                 const float* __restrict__ Wtopo, const float* __restrict__ btopo,
                 const float* __restrict__ pos,
                 float* __restrict__ x)
{
    const int b = blockIdx.x;
    const int tid = threadIdx.x;
    __shared__ float G[128 * 64];
    __shared__ float feat[1024];
    __shared__ float gbar[64];
    __shared__ float mconv[5];

    if (tid < 128) {          // per-diagram MLP + LN + gelu
        const float* p3 = pd + ((size_t)b * 128 + tid) * 3;
        const float p0 = p3[0], p1 = p3[1], p2 = p3[2];
        float t1[64]; float s = 0.f;
        for (int j = 0; j < 64; ++j) {
            const float v = p0 * Wpd1[j] + p1 * Wpd1[64 + j] + p2 * Wpd1[128 + j] + bpd1[j];
            t1[j] = v; s += v;
        }
        const float m = s * (1.f / 64.f);
        float var = 0.f;
        for (int j = 0; j < 64; ++j) { const float d = t1[j] - m; var += d * d; }
        const float inv = rsqrtf(var * (1.f / 64.f) + 1e-5f);
        for (int j = 0; j < 64; ++j)
            G[tid * 64 + j] = gelu_f((t1[j] - m) * inv * lnpdw[j] + lnpdb[j]);
    }
    __syncthreads();
    if (tid < 64) {           // mean over P (linearity: pool before Wpd2)
        float s = 0.f;
        for (int p = 0; p < 128; ++p) s += G[p * 64 + tid];
        gbar[tid] = s * (1.f / 128.f);
    } else if (tid >= 128 && tid < 133) {   // conv+mean collapses to 5 sums
        const int k = tid - 128;
        float s = 0.f;
        for (int xx = 0; xx < 128; ++xx) {
            const int p = 2 * xx - 2 + k;
            if (p >= 0 && p < 256) s += pim[(size_t)b * 256 + p];
        }
        mconv[k] = s * (1.f / 128.f);
    }
    __syncthreads();
    {
        float s = bpd2[tid];
        for (int j = 0; j < 64; ++j) s += gbar[j] * Wpd2[j * 256 + tid];
        feat[tid] = s;                                   // pd_pooled
        int i0 = betti[b * 3 + 0]; i0 = i0 < 0 ? 0 : (i0 > 9 ? 9 : i0);
        int i1 = betti[b * 3 + 1]; i1 = i1 < 0 ? 0 : (i1 > 9 ? 9 : i1);
        int i2 = betti[b * 3 + 2]; i2 = i2 < 0 ? 0 : (i2 > 9 ? 9 : i2);
        feat[256 + tid] = (btab[i0 * 256 + tid] + btab[i1 * 256 + tid] + btab[i2 * 256 + tid]) * (1.f / 3.f);
        #pragma unroll
        for (int e = 0; e < 2; ++e) {                    // piF: 512 channels
            const int c = tid * 2 + e;
            float a = convb[c];
            #pragma unroll
            for (int k = 0; k < 5; ++k) a += convw[c * 5 + k] * mconv[k];
            feat[512 + c] = a;
        }
    }
    __syncthreads();
    #pragma unroll
    for (int e = 0; e < 4; ++e) {                        // topo matvec
        const int n = tid * 4 + e;
        float a = btopo[n];
        for (int k = 0; k < 1024; ++k) a += feat[k] * Wtopo[(size_t)k * 1024 + n];
        x[((size_t)b * 512) * 1024 + n] = a + pos[n];
    }
}

__global__ __launch_bounds__(256)
void pool_kernel(const float* __restrict__ x, float* __restrict__ pooled)
{
    const int b = blockIdx.x;
    const int tid = threadIdx.x;
    for (int n = tid; n < 1024; n += 256) {
        float s = 0.f;
        for (int t = 0; t < 512; ++t) s += x[((size_t)(b * 512 + t)) * 1024 + n];
        pooled[b * 1024 + n] = s * (1.f / 512.f);
    }
}

__global__ __launch_bounds__(128)
void heads_kernel(const float* __restrict__ pooled,
                  const float* __restrict__ alnw, const float* __restrict__ alnb,
                  const float* __restrict__ Wa1, const float* __restrict__ ba1,
                  const float* __restrict__ Wa2, const float* __restrict__ ba2,
                  const float* __restrict__ clnw, const float* __restrict__ clnb,
                  const float* __restrict__ Wc1, const float* __restrict__ bc1,
                  const float* __restrict__ Wc2, const float* __restrict__ bc2,
                  float* __restrict__ out)
{
    const int b = blockIdx.x, tid = threadIdx.x;
    __shared__ float pr[1024];
    __shared__ float red[128];
    __shared__ float a1[64], c1[64], lg[10], smx[2];
    const float* src = pooled + (size_t)b * 1024;
    for (int c = tid; c < 1024; c += 128) pr[c] = src[c];
    __syncthreads();
    float s = 0.f;
    for (int c = tid; c < 1024; c += 128) s += pr[c];
    red[tid] = s; __syncthreads();
    for (int st = 64; st > 0; st >>= 1) { if (tid < st) red[tid] += red[tid + st]; __syncthreads(); }
    const float mean = red[0] * (1.f / 1024.f);
    __syncthreads();
    float v = 0.f;
    for (int c = tid; c < 1024; c += 128) { const float d = pr[c] - mean; v += d * d; }
    red[tid] = v; __syncthreads();
    for (int st = 64; st > 0; st >>= 1) { if (tid < st) red[tid] += red[tid + st]; __syncthreads(); }
    const float inv = rsqrtf(red[0] * (1.f / 1024.f) + 1e-5f);
    __syncthreads();
    if (tid < 64) {
        float acc = ba1[tid];
        for (int k = 0; k < 1024; ++k)
            acc += ((pr[k] - mean) * inv * alnw[k] + alnb[k]) * Wa1[k * 64 + tid];
        a1[tid] = gelu_f(acc);
    } else {
        const int t2 = tid - 64;
        float acc = bc1[t2];
        for (int k = 0; k < 1024; ++k)
            acc += ((pr[k] - mean) * inv * clnw[k] + clnb[k]) * Wc1[k * 64 + t2];
        c1[t2] = gelu_f(acc);
    }
    __syncthreads();
    if (tid == 0) {
        float a2 = ba2[0];
        for (int t = 0; t < 64; ++t) a2 += a1[t] * Wa2[t];
        out[b] = 1.f / (1.f + expf(-a2));
    }
    if (tid >= 16 && tid < 26) {
        const int j = tid - 16;
        float acc = bc2[j];
        for (int t = 0; t < 64; ++t) acc += c1[t] * Wc2[t * 10 + j];
        lg[j] = acc;
        out[32 + b * 10 + j] = acc;
    }
    __syncthreads();
    if (tid == 0) {
        float mx = lg[0];
        for (int j = 1; j < 10; ++j) mx = fmaxf(mx, lg[j]);
        float se = 0.f;
        for (int j = 0; j < 10; ++j) se += expf(lg[j] - mx);
        smx[0] = mx; smx[1] = 1.f / se;
    }
    __syncthreads();
    if (tid < 10) out[352 + b * 10 + tid] = expf(lg[tid] - smx[0]) * smx[1];
}

__global__ void f32_to_f16_kernel(const float* __restrict__ in, _Float16* __restrict__ out, long n)
{
    long i = (long)blockIdx.x * 256 + threadIdx.x;
    const long stride = (long)gridDim.x * 256;
    for (; i < n; i += stride) out[i] = (_Float16)in[i];
}

// ---------------------------------------------------------------------------
extern "C" void kernel_launch(void* const* d_in, const int* in_sizes, int n_in,
                              void* d_out, int out_size, void* d_ws, size_t ws_size,
                              hipStream_t stream)
{
    const float* pd    = (const float*)d_in[0];
    const int*   betti = (const int*)  d_in[1];
    const float* pim   = (const float*)d_in[2];
    const float* seqf  = (const float*)d_in[3];
    const float* Wpd1  = (const float*)d_in[4];
    const float* bpd1  = (const float*)d_in[5];
    const float* lnpdw = (const float*)d_in[6];
    const float* lnpdb = (const float*)d_in[7];
    const float* Wpd2  = (const float*)d_in[8];
    const float* bpd2  = (const float*)d_in[9];
    const float* btab  = (const float*)d_in[10];
    const float* convw = (const float*)d_in[11];
    const float* convb = (const float*)d_in[12];
    const float* Wtopo = (const float*)d_in[13];
    const float* btopo = (const float*)d_in[14];
    const float* W_seq = (const float*)d_in[15];
    const float* b_seq = (const float*)d_in[16];
    const float* pos   = (const float*)d_in[17];
    const float* Wqkv  = (const float*)d_in[18];
    const float* Wo    = (const float*)d_in[19];
    const float* bo    = (const float*)d_in[20];
    const float* Rmat  = (const float*)d_in[21];
    const float* ln1w  = (const float*)d_in[22];
    const float* ln1b  = (const float*)d_in[23];
    const float* ln2w  = (const float*)d_in[24];
    const float* ln2b  = (const float*)d_in[25];
    const float* Wff1  = (const float*)d_in[26];
    const float* bff1  = (const float*)d_in[27];
    const float* Wff2  = (const float*)d_in[28];
    const float* bff2  = (const float*)d_in[29];
    const float* alnw  = (const float*)d_in[30];
    const float* alnb  = (const float*)d_in[31];
    const float* Wa1   = (const float*)d_in[32];
    const float* ba1   = (const float*)d_in[33];
    const float* Wa2   = (const float*)d_in[34];
    const float* ba2   = (const float*)d_in[35];
    const float* clnw  = (const float*)d_in[36];
    const float* clnb  = (const float*)d_in[37];
    const float* Wc1   = (const float*)d_in[38];
    const float* bc1   = (const float*)d_in[39];
    const float* Wc2   = (const float*)d_in[40];
    const float* bc2   = (const float*)d_in[41];
    float* out = (float*)d_out;
    (void)in_sizes; (void)n_in; (void)out_size; (void)ws_size;

    // workspace layout (~228 MB)
    char* ws = (char*)d_ws;
    size_t off = 0;
    float*    x      = (float*)(ws + off);    off += (size_t)16384 * 1024 * 4;  // residual stream
    _Float16* hbuf   = (_Float16*)(ws + off); off += (size_t)16384 * 1024 * 2;  // LN out / z (f16)
    _Float16* qkvbuf = (_Float16*)(ws + off); off += (size_t)16384 * 3072 * 2;  // qkv / ff1 (f16)
    _Float16* qpbuf  = (_Float16*)(ws + off); off += (size_t)16384 * 512 * 2;
    _Float16* kpbuf  = (_Float16*)(ws + off); off += (size_t)16384 * 512 * 2;
    float*    kvbuf  = (float*)(ws + off);    off += (size_t)512 * 2048 * 4;
    float*    ksbuf  = (float*)(ws + off);    off += (size_t)512 * 32 * 4;
    float*    pooled = (float*)(ws + off);    off += (size_t)32 * 1024 * 4;

    // topo row (x[:,0,:])
    topo_kernel<<<32, 256, 0, stream>>>(pd, betti, pim, Wpd1, bpd1, lnpdw, lnpdb,
                                        Wpd2, bpd2, btab, convw, convb, Wtopo, btopo, pos, x);

    // sequence embedding rows (x[:,1:,:]) via WMMA
    const long nseq = (long)32 * 511 * 128;
    f32_to_f16_kernel<<<2048, 256, 0, stream>>>(seqf, hbuf, nseq);
    gemm_wmma_kernel<EPI_EMBED><<<dim3(8, 128), 256, 0, stream>>>(
        hbuf, W_seq, b_seq, (void*)x, nullptr, pos, 16352, 1024, 128);

    for (int l = 0; l < 6; ++l) {
        ln_rows_kernel<<<16384, 128, 0, stream>>>(x, ln1w + l * 1024, ln1b + l * 1024, hbuf);
        gemm_wmma_kernel<EPI_F16><<<dim3(24, 128), 256, 0, stream>>>(
            hbuf, Wqkv + (size_t)l * 1024 * 3072, nullptr, (void*)qkvbuf, nullptr, nullptr,
            16384, 3072, 1024);
        qpkp_kernel<<<16384, 256, 0, stream>>>(qkvbuf, Rmat + l * 64 * 32, qpbuf, kpbuf);
        kvsum_kernel<<<512, 256, 0, stream>>>(kpbuf, qkvbuf, kvbuf, ksbuf);
        z_kernel<<<2048, 256, 0, stream>>>(qpbuf, kvbuf, ksbuf, hbuf);
        gemm_wmma_kernel<EPI_RES_F32><<<dim3(8, 128), 256, 0, stream>>>(
            hbuf, Wo + (size_t)l * 1024 * 1024, bo + l * 1024, nullptr, x, nullptr,
            16384, 1024, 1024);
        ln_rows_kernel<<<16384, 128, 0, stream>>>(x, ln2w + l * 1024, ln2b + l * 1024, hbuf);
        gemm_wmma_kernel<EPI_F16_GELU><<<dim3(16, 128), 256, 0, stream>>>(
            hbuf, Wff1 + (size_t)l * 1024 * 2048, bff1 + l * 2048, (void*)qkvbuf, nullptr, nullptr,
            16384, 2048, 1024);
        gemm_wmma_kernel<EPI_RES_F32><<<dim3(8, 128), 256, 0, stream>>>(
            qkvbuf, Wff2 + (size_t)l * 2048 * 1024, bff2 + l * 1024, nullptr, x, nullptr,
            16384, 1024, 2048);
    }

    pool_kernel<<<32, 256, 0, stream>>>(x, pooled);
    heads_kernel<<<32, 128, 0, stream>>>(pooled, alnw, alnb, Wa1, ba1, Wa2, ba2,
                                         clnw, clnb, Wc1, bc1, Wc2, bc2, out);
}
